// SkeletalEncBlock_70325794504755
// MI455X (gfx1250) — matrix operand order repaired and checked by reading, not compiled, not run on hardware
//
#include <hip/hip_runtime.h>

typedef __attribute__((ext_vector_type(16))) __bf16 v16bf;
typedef __attribute__((ext_vector_type(8)))  float  v8f;

#define NJ 24
#define NCI 32
#define NCO 32
#define NT 4096
#define NB 8
#define NR 12
#define TT 32                 // time tile per workgroup
#define ROWE 776              // 768 channels + 8 pad bf16 elems per LDS row (1552 B, 16B aligned)
#define NEDGE 70
#define WFRAG_BYTES (NEDGE * 3 * 2 * 32 * 32)   // 430,080 B of bf16 A-fragments
#define POOLED_ELEMS (NB * NR * NCO * NT)       // 12,582,912 floats

// Skeleton adjacency (flattened), per-joint degree, edge offsets, edge->joint map.
__device__ __constant__ int c_deg[NJ]  = {4,3,3,3,3,3,3,3,3,5,2,2,3,3,3,2,3,3,3,3,3,3,2,2};
__device__ __constant__ int c_eoff[NJ] = {0,4,7,10,13,16,19,22,25,28,33,35,37,40,43,46,48,51,54,57,60,63,66,68};
__device__ __constant__ int c_adj[NEDGE] = {
  0,1,2,3,  1,0,4,  2,0,5,  3,0,6,  4,1,7,  5,2,8,  6,3,9,  7,4,10,  8,5,11,
  9,6,12,13,14,  10,7,  11,8,  12,9,15,  13,9,16,  14,9,17,  15,12,
  16,13,18,  17,14,19,  18,16,20,  19,17,21,  20,18,22,  21,19,23,  22,20,  23,21 };
__device__ __constant__ int c_ej[NEDGE] = {
  0,0,0,0, 1,1,1, 2,2,2, 3,3,3, 4,4,4, 5,5,5, 6,6,6, 7,7,7, 8,8,8,
  9,9,9,9,9, 10,10, 11,11, 12,12,12, 13,13,13, 14,14,14, 15,15,
  16,16,16, 17,17,17, 18,18,18, 19,19,19, 20,20,20, 21,21,21, 22,22, 23,23 };

// fp32 -> bf16 bits, round-to-nearest-even
__device__ __forceinline__ unsigned int f2bfu(float f) {
  unsigned int u = __float_as_uint(f);
  unsigned int lsb = (u >> 16) & 1u;
  u += 0x7fffu + lsb;
  return u >> 16;
}

// ---------------------------------------------------------------------------
// Kernel 1: build masked, bf16, A-fragment-swizzled weights in workspace.
// Layout: wf[((e*3 + tap)*2 + half)*512 + lane*16 + q]
// A-matrix 16x32 bf16: lane<16 -> K {0..15}, lane>=16 -> K {16..31} (matches B fill)
// ---------------------------------------------------------------------------
__global__ void skel_prep_wfrag(const float* __restrict__ W,
                                unsigned short* __restrict__ wf) {
  int gid = blockIdx.x * blockDim.x + threadIdx.x;
  if (gid >= NEDGE * 3 * 2 * 32) return;
  int lane = gid & 31;
  int half = (gid >> 5) & 1;
  int tap  = (gid >> 6) % 3;
  int e    = gid / 192;
  int j = c_ej[e];
  int i = c_adj[e];
  int co = half * 16 + (lane & 15);
  unsigned short* dst = wf + ((size_t)((e * 3 + tap) * 2 + half)) * 512 + lane * 16;
#pragma unroll
  for (int q = 0; q < 16; ++q) {
    int p = q >> 1, h = q & 1;
    int cin = ((lane & 16) ? 16 : 0) + (p << 1) + h;
    float w = W[((size_t)(j * NCO + co) * (NJ * NCI) + (size_t)(i * NCI + cin)) * 3 + tap];
    dst[q] = (unsigned short)f2bfu(w);
  }
}

// ---------------------------------------------------------------------------
// Kernel 2: masked conv (bf16 WMMA, f32 accum) + bias + leakyReLU + pair-pool.
// One workgroup per (batch, 32-wide time tile). 8 waves, 24 (region, t16) tasks.
// ---------------------------------------------------------------------------
__global__ void __launch_bounds__(256)
skel_conv_wmma(const float* __restrict__ x, const float* __restrict__ bias,
               const unsigned short* __restrict__ wf, float* __restrict__ out) {
  __shared__ __align__(16) unsigned short xs[(TT + 2) * ROWE];  // 52.8 KB, bf16 [t][ch]

  const int tid  = threadIdx.x;
  const int bIdx = blockIdx.y;
  const int t0   = blockIdx.x * TT;

  // Warm L2 with the weight-fragment buffer (reused by every workgroup).
  for (size_t off = (size_t)tid * 128; off < WFRAG_BYTES; off += 256 * 128)
    __builtin_prefetch((const char*)wf + off, 0, 1);

  // Stage x tile (with +/-1 halo for SAME conv) into LDS, transposed & bf16.
  // Two adjacent channels are packed into one b32 LDS store.
  const float* xb = x + (size_t)bIdx * (NJ * NCI) * NT;
  if (t0 > 0 && t0 + TT < NT) {
    // interior: no boundary predication
    for (int idx = tid; idx < 384 * (TT + 2); idx += 256) {
      int chp = idx / (TT + 2);
      int tl  = idx - chp * (TT + 2);
      int tg  = t0 + tl - 1;
      float v0 = xb[(size_t)(2 * chp) * NT + tg];
      float v1 = xb[(size_t)(2 * chp + 1) * NT + tg];
      unsigned int pk = f2bfu(v0) | (f2bfu(v1) << 16);
      *(unsigned int*)&xs[tl * ROWE + 2 * chp] = pk;
    }
  } else {
    for (int idx = tid; idx < 384 * (TT + 2); idx += 256) {
      int chp = idx / (TT + 2);
      int tl  = idx - chp * (TT + 2);
      int tg  = t0 + tl - 1;
      bool ok = (tg >= 0) && (tg < NT);
      float v0 = ok ? xb[(size_t)(2 * chp) * NT + tg] : 0.0f;
      float v1 = ok ? xb[(size_t)(2 * chp + 1) * NT + tg] : 0.0f;
      unsigned int pk = f2bfu(v0) | (f2bfu(v1) << 16);
      *(unsigned int*)&xs[tl * ROWE + 2 * chp] = pk;
    }
  }
  __syncthreads();

  const int wave  = tid >> 5;
  const int lane  = tid & 31;
  const int n     = lane & 15;              // N (time) index within 16-tile
  const int hi    = (lane & 16) ? 1 : 0;
  const int kbase = hi ? 16 : 0;            // B-matrix K striping across lane halves
  const v8f zero = {0.f, 0.f, 0.f, 0.f, 0.f, 0.f, 0.f, 0.f};

  for (int task = wave; task < 24; task += 8) {
    const int r = task % 12;                // region (joint pair)
    const int s = task / 12;                // 16-wide time sub-tile
    v8f acc[2][2] = {{zero, zero}, {zero, zero}};   // [joint-in-pair][co-half]

#pragma unroll
    for (int jj = 0; jj < 2; ++jj) {
      const int j  = 2 * r + jj;
      const int eo = c_eoff[j];
      const int dg = c_deg[j];
      for (int slot = 0; slot < dg; ++slot) {
        const int e   = eo + slot;
        const int nbj = c_adj[e];
#pragma unroll
        for (int tap = 0; tap < 3; ++tap) {
          const int row = s * 16 + n + tap; // LDS row == global t0-1+row
          v16bf bfrag = *(const v16bf*)&xs[row * ROWE + nbj * NCI + kbase];
          const unsigned short* wb = wf + ((size_t)(e * 3 + tap)) * 1024 + lane * 16;
          v16bf a0 = *(const v16bf*)(wb);
          v16bf a1 = *(const v16bf*)(wb + 512);
          acc[jj][0] = __builtin_amdgcn_wmma_f32_16x16x32_bf16(
              false, a0, false, bfrag, (short)0, acc[jj][0], false, false);
          acc[jj][1] = __builtin_amdgcn_wmma_f32_16x16x32_bf16(
              false, a1, false, bfrag, (short)0, acc[jj][1], false, false);
        }
      }
    }

    // Epilogue: bias -> leaky ReLU -> mean over the joint pair -> store.
#pragma unroll
    for (int half = 0; half < 2; ++half) {
      v8f p = zero;
#pragma unroll
      for (int jj = 0; jj < 2; ++jj) {
        const int j = 2 * r + jj;
#pragma unroll
        for (int q = 0; q < 8; ++q) {
          float v = acc[jj][half][q] + bias[j * NCO + half * 16 + q + hi * 8];
          v = (v > 0.0f) ? v : 0.2f * v;
          p[q] += 0.5f * v;
        }
      }
      // C/D layout: vgpr q -> M = half*16 + q + hi*8 ; lane%16 -> N (time)
      size_t obase = (((size_t)bIdx * (NR * NCO)) + (size_t)(r * NCO + half * 16 + hi * 8)) * NT
                     + (size_t)(t0 + s * 16 + n);
#pragma unroll
      for (int q = 0; q < 8; ++q)
        out[obase + (size_t)q * NT] = p[q];
    }
  }
}

// ---------------------------------------------------------------------------
// Kernel 3: tiny metadata outputs (REGIONS then POST_ADJ) at tail of d_out.
// ---------------------------------------------------------------------------
__global__ void skel_write_meta(int* __restrict__ meta) {
  int tid = threadIdx.x;
  if (tid < 24) meta[tid] = tid;                 // REGIONS = arange(24).reshape(12,2)
  if (tid < 144) {
    int r1 = tid / 12, r2 = tid % 12;
    int v = 0;
#pragma unroll
    for (int a = 0; a < 2; ++a) {
      int ja = 2 * r1 + a;
      int eo = c_eoff[ja], dg = c_deg[ja];
      for (int s2 = 0; s2 < dg; ++s2)
        if ((c_adj[eo + s2] >> 1) == r2) v = 1;
    }
    meta[24 + tid] = v;                          // POST_ADJ[r1][r2]
  }
}

extern "C" void kernel_launch(void* const* d_in, const int* in_sizes, int n_in,
                              void* d_out, int out_size, void* d_ws, size_t ws_size,
                              hipStream_t stream) {
  const float* x = (const float*)d_in[0];   // (8, 768, 4096) f32
  const float* W = (const float*)d_in[1];   // (768, 768, 3)  f32
  const float* b = (const float*)d_in[2];   // (768,)         f32
  float* out = (float*)d_out;
  unsigned short* wf = (unsigned short*)d_ws;   // bf16 weight fragments

  const int prepThreads = NEDGE * 3 * 2 * 32;   // 13,440
  skel_prep_wfrag<<<(prepThreads + 255) / 256, 256, 0, stream>>>(W, wf);

  dim3 grid(NT / TT, NB);                        // 128 x 8 workgroups
  skel_conv_wmma<<<grid, 256, 0, stream>>>(x, b, wf, out);

  skel_write_meta<<<1, 256, 0, stream>>>((int*)(out + POOLED_ELEMS));
}
